// S4_76347338654366
// MI455X (gfx1250) — matrix operand order repaired and checked by reading, not compiled, not run on hardware
//
#include <hip/hip_runtime.h>
#include <hip/hip_bf16.h>

// Problem constants (match reference)
#define Bn 64
#define Tn 2048
#define Hn 512
#define Vn 65
#define On 65
#define Ln 3
#define HH (Hn * Hn)
#define LDSP (Hn + 8)   // padded LDS row stride (bf16 elems) -> conflict-free ds_load_b128

typedef __attribute__((ext_vector_type(16))) __bf16 v16bf;
typedef __attribute__((ext_vector_type(8)))  float  v8f;
typedef __attribute__((ext_vector_type(4)))  int    v4i_;

using bf = __hip_bfloat16;

union Frag {
  uint4 q[2];
  v16bf v;
};

#if __has_builtin(__builtin_amdgcn_global_load_async_to_lds_b128)
#define HAVE_ASYNC_LDS 1
#else
#define HAVE_ASYNC_LDS 0
#endif

__device__ __forceinline__ void wait_asynccnt0() {
#if __has_builtin(__builtin_amdgcn_s_wait_asynccnt)
  __builtin_amdgcn_s_wait_asynccnt(0);
#else
  asm volatile("s_wait_asynccnt 0" ::: "memory");
#endif
}

// 16B global -> LDS copy; async path on CDNA5 (ASYNCcnt-tracked, bypasses VGPRs),
// VGPR bounce otherwise.
__device__ __forceinline__ void cp16_g2l(const bf* gsrc, bf* ldst) {
#if HAVE_ASYNC_LDS
  __builtin_amdgcn_global_load_async_to_lds_b128(
      (__attribute__((address_space(1))) v4i_*)gsrc,
      (__attribute__((address_space(3))) v4i_*)ldst, 0, 0);
#else
  *(uint4*)ldst = *(const uint4*)gsrc;
#endif
}

// ---- WMMA helpers -------------------------------------------------------
// A-operand 16x32 bf16: lane m = lane&15 holds row m; lanes 0-15 K={k..k+7, k+16..k+23},
// lanes 16-31 K={k+8..k+15, k+24..k+31}  -> two 16B loads at +hi*8 and +16+hi*8
__device__ __forceinline__ v16bf load_a_frag(const bf* rowp, int k, int hi) {
  Frag f;
  const bf* p = rowp + k + hi * 8;
  f.q[0] = *(const uint4*)(p);
  f.q[1] = *(const uint4*)(p + 16);
  return f.v;
}
// B-operand 32x16 bf16: lane n = lane&15 holds column n; lanes 0-15 K={k..k+7, k+16..k+23},
// lanes 16-31 K={k+8..k+15, k+24..k+31} -> two 16B loads at +hi*8 and +16+hi*8
__device__ __forceinline__ v16bf load_b_frag(const bf* rowp, int k, int hi) {
  Frag f;
  const bf* p = rowp + k + hi * 8;
  f.q[0] = *(const uint4*)(p);
  f.q[1] = *(const uint4*)(p + 16);
  return f.v;
}

__device__ __forceinline__ v8f wmma_bf16(v16bf a, v16bf b, v8f c) {
  // (neg_a, A, neg_b, B, c_mod, C, reuse_a, reuse_b)
  return __builtin_amdgcn_wmma_f32_16x16x32_bf16(false, a, false, b, (short)0, c,
                                                 false, false);
}

// Branch-free tanh: V_TANH_F32 if the builtin exists, else exp2/rcp formulation
// (saturates correctly to +/-1 at the limits; no EXEC divergence).
__device__ __forceinline__ float fast_tanh(float y) {
#if __has_builtin(__builtin_amdgcn_tanhf)
  return __builtin_amdgcn_tanhf(y);
#elif __has_builtin(__builtin_amdgcn_tanh_f32)
  return __builtin_amdgcn_tanh_f32(y);
#else
  // tanh(y) = 1 - 2/(1 + exp2(y * 2*log2(e)))
  float e = __builtin_amdgcn_exp2f(y * 2.8853900817779268f);
  return 1.0f - 2.0f * __builtin_amdgcn_rcpf(e + 1.0f);
#endif
}

__device__ __forceinline__ float gelu_f(float x) {
  return 0.5f * x * (1.0f + fast_tanh(0.7978845608028654f * (x + 0.044715f * x * x * x)));
}

// ---- Per-strip sense-reversing barrier (8 workgroups, L2-resident flags) ----
__device__ __forceinline__ void strip_sync(unsigned* cnt, unsigned* flag,
                                           unsigned nwg, unsigned ph) {
  __threadfence();          // make this thread's global writes visible (device scope)
  __syncthreads();
  if (threadIdx.x == 0) {
    unsigned prev = __hip_atomic_fetch_add(cnt, 1u, __ATOMIC_ACQ_REL,
                                           __HIP_MEMORY_SCOPE_AGENT);
    if (prev == nwg - 1u) {
      __hip_atomic_store(cnt, 0u, __ATOMIC_RELAXED, __HIP_MEMORY_SCOPE_AGENT);
      __hip_atomic_store(flag, ph, __ATOMIC_RELEASE, __HIP_MEMORY_SCOPE_AGENT);
    } else {
      while (__hip_atomic_load(flag, __ATOMIC_ACQUIRE, __HIP_MEMORY_SCOPE_AGENT) < ph) {
        __builtin_amdgcn_s_sleep(1);
      }
    }
  }
  __syncthreads();
}

// ---- Kernel 0: convert/transpose weights to bf16 (L2-resident thereafter) ----
__global__ void prep_kernel(const float* __restrict__ W_in, const float* __restrict__ A,
                            const float* __restrict__ Bm, const float* __restrict__ W_out,
                            const float* __restrict__ emb,
                            bf* __restrict__ Wn, bf* __restrict__ Abf,
                            bf* __restrict__ BbfT, bf* __restrict__ Wo,
                            bf* __restrict__ embb) {
  size_t i = (size_t)blockIdx.x * blockDim.x + threadIdx.x;
  const size_t r0 = HH;
  const size_t r1 = r0 + (size_t)Ln * HH;
  const size_t r2 = r1 + (size_t)Ln * HH;
  const size_t r3 = r2 + (size_t)80 * Hn;
  const size_t r4 = r3 + (size_t)Vn * Hn;
  if (i < r0) {
    Wn[i] = __float2bfloat16(W_in[i]);                 // rows = out col, K contiguous
  } else if (i < r1) {
    size_t j = i - r0;
    Abf[j] = __float2bfloat16(A[j]);                   // h@A.T: A rows = out col (K contig)
  } else if (i < r2) {
    size_t j = i - r1;
    size_t l = j / HH, r = j % HH;
    size_t nn = r / Hn, kk = r % Hn;
    BbfT[j] = __float2bfloat16(Bm[l * HH + kk * Hn + nn]);  // transpose: [n][k]
  } else if (i < r3) {
    size_t j = i - r2;                                 // W_out padded 65 -> 80 rows
    Wo[j] = (j < (size_t)On * Hn) ? __float2bfloat16(W_out[j]) : __float2bfloat16(0.0f);
  } else if (i < r4) {
    size_t j = i - r3;
    embb[j] = __float2bfloat16(emb[j]);
  }
}

// ---- Kernel 1: zero h-state + barrier flags (every launch; ws is poisoned) ----
__global__ void init_kernel(unsigned* __restrict__ h_words, unsigned* __restrict__ sync_words) {
  int i = blockIdx.x * blockDim.x + threadIdx.x;
  const int HW = Ln * Bn * Hn / 2;   // bf16 state as u32 words
  if (i < HW) h_words[i] = 0u;
  if (i < 128) sync_words[i] = 0u;
}

// ---- Kernel 2: x = bf16(gelu(emb[tok] @ W_in^T + b_in)), whole chip ----------
__global__ __launch_bounds__(128) void embed_kernel(const int* __restrict__ xseq,
                                                    const bf* __restrict__ embb,
                                                    const bf* __restrict__ Wn,
                                                    const float* __restrict__ b_in,
                                                    bf* __restrict__ Xseq) {
  const int lane = threadIdx.x & 31, wave = threadIdx.x >> 5;
  const int m0 = blockIdx.x * 16;
  const int n0 = blockIdx.y * 64 + wave * 16;
  const int m = lane & 15, hi = lane >> 4, nn = lane & 15;
  const int tok = xseq[m0 + m];
  const bf* arow = embb + (size_t)tok * Hn;
  const bf* brow = Wn + (size_t)(n0 + nn) * Hn;
  v8f acc = {};
#pragma unroll 4
  for (int k = 0; k < Hn; k += 32) {
    acc = wmma_bf16(load_a_frag(arow, k, hi), load_b_frag(brow, k, hi), acc);
  }
  const int col = n0 + nn;
  const float bi = b_in[col];
#pragma unroll
  for (int r = 0; r < 8; ++r) {
    int row = m0 + r + hi * 8;   // C-layout: VGPR r -> M = r (lanes<16) / r+8 (lanes>=16)
    Xseq[(size_t)row * Hn + col] = __float2bfloat16(gelu_f(acc[r] + bi));
  }
}

// ---- Kernel 3: persistent SSM scan ------------------------------------------
// grid = (8 col-WGs, 4 batch strips), block = 128 (4 waves; wave = 16-col strip)
__global__ __launch_bounds__(128) void scan_kernel(const bf* __restrict__ Xseq,
                                                   const float* __restrict__ b_out,
                                                   bf* __restrict__ hbf, bf* __restrict__ xbf,
                                                   const bf* __restrict__ Abf,
                                                   const bf* __restrict__ BbfT,
                                                   const bf* __restrict__ Wo,
                                                   float* __restrict__ out,
                                                   unsigned* __restrict__ sync) {
  __shared__ bf hsh[16 * LDSP];
  __shared__ bf xsh[16 * LDSP];

  const int lane = threadIdx.x & 31, wave = threadIdx.x >> 5;
  const int strip = blockIdx.y;
  const int m0 = strip * 16;                       // batch rows of this strip
  const int ncol0 = (blockIdx.x * 4 + wave) * 16;  // output columns of this wave
  const int m = lane & 15, hi = lane >> 4, nn = lane & 15;

  unsigned* cnt  = sync + strip * 32;
  unsigned* flag = sync + strip * 32 + 16;
  const unsigned nwg = gridDim.x;

  // per-lane weight row pointers (constant over time; stay hot in L2)
  const bf* wA[Ln];
  const bf* wB[Ln];
#pragma unroll
  for (int l = 0; l < Ln; ++l) {
    wA[l] = Abf  + (size_t)l * HH + (size_t)(ncol0 + nn) * Hn;
    wB[l] = BbfT + (size_t)l * HH + (size_t)(ncol0 + nn) * Hn;
  }

  // loop-invariant output-projection setup (col-WG 0 only): wave w -> tile w,
  // wave 0 additionally covers padded tile 4 (cols 64..79, only col 64 live)
  const bf* x2row = xbf + (size_t)2 * Bn * Hn + (size_t)(m0 + m) * Hn;
  const int  npass = (wave == 0) ? 2 : 1;
  const bf*  wrowP[2];
  int        colP[2];
  float      boP[2];
#pragma unroll
  for (int pass = 0; pass < 2; ++pass) {
    int tile = (pass == 0) ? wave : 4;
    wrowP[pass] = Wo + (size_t)(tile * 16 + nn) * Hn;
    colP[pass]  = tile * 16 + nn;
    boP[pass]   = (colP[pass] < On) ? b_out[colP[pass]] : 0.0f;
  }

  unsigned ph = 0;
  for (int t = 0; t < Tn; ++t) {
#pragma unroll 1
    for (int l = 0; l < Ln; ++l) {
      // Stage h-panel and x-panel (16 rows x 512 cols, bf16) into LDS via async
      // global->LDS b128 copies (ASYNCcnt), shared by the block's 4 waves.
      {
        const bf* hsrc = hbf + (size_t)l * Bn * Hn + (size_t)m0 * Hn;
        const bf* xsrc;
        size_t xstride;
        if (l == 0) { xsrc = Xseq + ((size_t)m0 * Tn + (size_t)t) * Hn; xstride = (size_t)Tn * Hn; }
        else        { xsrc = xbf + (size_t)(l - 1) * Bn * Hn + (size_t)m0 * Hn; xstride = Hn; }
        for (int c = threadIdx.x; c < 16 * (Hn / 8); c += 128) {
          int row = c >> 6;
          int off = (c & 63) * 8;
          cp16_g2l(hsrc + (size_t)row * Hn + off, &hsh[row * LDSP + off]);
          const bf* xg = xsrc + (size_t)row * xstride + off;
          cp16_g2l(xg, &xsh[row * LDSP + off]);
          if (l == 0 && t + 1 < Tn) {
            // next timestep's Xseq panel is the only stream not resident in L2
            __builtin_prefetch((const void*)(xg + Hn), 0, 1);
          }
        }
        wait_asynccnt0();
      }
      __syncthreads();

      v8f acc = {};
      const bf* hrow = hsh + m * LDSP;
      const bf* xrow = xsh + m * LDSP;
#pragma unroll 4
      for (int k = 0; k < Hn; k += 32) {
        acc = wmma_bf16(load_a_frag(hrow, k, hi), load_b_frag(wA[l], k, hi), acc);
        acc = wmma_bf16(load_a_frag(xrow, k, hi), load_b_frag(wB[l], k, hi), acc);
      }

      bf* hout = hbf + (size_t)l * Bn * Hn;
      bf* xout = xbf + (size_t)l * Bn * Hn;
      const int colH = ncol0 + nn;
#pragma unroll
      for (int r = 0; r < 8; ++r) {
        int brow = m0 + r + hi * 8;
        float v = acc[r];
        hout[(size_t)brow * Hn + colH] = __float2bfloat16(v);          // bf16 state carry
        xout[(size_t)brow * Hn + colH] = __float2bfloat16(gelu_f(v));  // next layer input
      }

      ++ph;
      strip_sync(cnt, flag, nwg, ph);
    }

    // Output projection y[b,t,:] = gelu(h2) @ W_out^T + b_out, by col-WG 0 of each strip.
    if (blockIdx.x == 0) {
      for (int pass = 0; pass < npass; ++pass) {
        const bf* wrow = wrowP[pass];
        v8f acc = {};
#pragma unroll 4
        for (int k = 0; k < Hn; k += 32) {
          acc = wmma_bf16(load_a_frag(x2row, k, hi), load_b_frag(wrow, k, hi), acc);
        }
        if (colP[pass] < On) {
          const float bo = boP[pass];
          const int col = colP[pass];
#pragma unroll
          for (int r = 0; r < 8; ++r) {
            int brow = m0 + r + hi * 8;
            out[((size_t)brow * Tn + (size_t)t) * On + col] = acc[r] + bo;
          }
        }
      }
    }
  }
}

// ---- Host launch ------------------------------------------------------------
extern "C" void kernel_launch(void* const* d_in, const int* in_sizes, int n_in,
                              void* d_out, int out_size, void* d_ws, size_t ws_size,
                              hipStream_t stream) {
  (void)in_sizes; (void)n_in; (void)out_size; (void)ws_size;
  const int*   x_seq = (const int*)  d_in[0];
  const float* emb   = (const float*)d_in[1];
  const float* W_in  = (const float*)d_in[2];
  const float* b_in  = (const float*)d_in[3];
  const float* A     = (const float*)d_in[4];
  const float* Bm    = (const float*)d_in[5];
  const float* W_out = (const float*)d_in[6];
  const float* b_out = (const float*)d_in[7];
  float* out = (float*)d_out;

  char* ws = (char*)d_ws;
  size_t off = 0;
  auto take = [&](size_t bytes) -> char* {
    char* p = ws + off;
    off = (off + bytes + 255) & ~(size_t)255;
    return p;
  };
  bf* Wn   = (bf*)take((size_t)HH * 2);
  bf* Abf  = (bf*)take((size_t)Ln * HH * 2);
  bf* BbfT = (bf*)take((size_t)Ln * HH * 2);
  bf* Wo   = (bf*)take((size_t)80 * Hn * 2);
  bf* embb = (bf*)take((size_t)Vn * Hn * 2);
  bf* Xseq = (bf*)take((size_t)Bn * Tn * Hn * 2);   // 128 MB bf16 activations
  bf* hbf  = (bf*)take((size_t)Ln * Bn * Hn * 2);
  bf* xbf  = (bf*)take((size_t)Ln * Bn * Hn * 2);
  unsigned* sync = (unsigned*)take(4096);

  const size_t preptotal = (size_t)HH + 2 * (size_t)Ln * HH + (size_t)80 * Hn + (size_t)Vn * Hn;
  prep_kernel<<<dim3((unsigned)((preptotal + 255) / 256)), 256, 0, stream>>>(
      W_in, A, Bm, W_out, emb, Wn, Abf, BbfT, Wo, embb);

  const int init_words = Ln * Bn * Hn / 2;
  init_kernel<<<dim3((init_words + 255) / 256), 256, 0, stream>>>((unsigned*)hbf, sync);

  embed_kernel<<<dim3(Bn * Tn / 16, Hn / 64), 128, 0, stream>>>(x_seq, embb, Wn, b_in, Xseq);

  scan_kernel<<<dim3(8, 4), 128, 0, stream>>>(Xseq, b_out, hbf, xbf, Abf, BbfT, Wo, out, sync);
}
